// PairwiseInteractions_12111807774822
// MI455X (gfx1250) — matrix operand order repaired and checked by reading, not compiled, not run on hardware
//
#include <hip/hip_runtime.h>

// PairwiseInteractions on gfx1250 (MI455X, wave32):
//   Kernel 1: LDS-staged O(N^2) KNN (top-17 incl. self) with register insertion list.
//   Kernel 2: one wave per point; 16-neighbor x 12-feature x 64-output MLP via
//             v_wmma_f32_16x16x32_f16 (K padded 12->32), bias in accumulator,
//             ReLU + row reduction via shfl_xor(16).

#define NPOINTS 4096
#define DOUT    64
#define NNBR    16      // neighbors used (reference drops self from top-17)
#define KLIST   17

typedef __attribute__((ext_vector_type(16))) _Float16 v16h;
typedef __attribute__((ext_vector_type(8)))  float    v8f;

// ---------------------------------------------------------------------------
// Kernel 1: per-batch KNN. Block = 256 threads = 256 query points.
// Whole batch's coords + |x|^2 staged in LDS (4096 * float4 = 64 KB, fits WGP LDS).
// Each thread scans all 4096 candidates keeping a sorted 17-entry (d2, idx)
// list in registers (fully unrolled percolation insert -> no scratch).
// Ordering matches jax top_k(-dist): ascending d2, ties broken by lower index.
// ---------------------------------------------------------------------------
__global__ __launch_bounds__(256) void knn_kernel(const float* __restrict__ x0,
                                                  int* __restrict__ nbr) {
  __shared__ float4 sx[NPOINTS];
  const int b = blockIdx.y;
  const float* xb = x0 + (size_t)b * NPOINTS * 3;

  for (int p = threadIdx.x; p < NPOINTS; p += blockDim.x) {
    float x = xb[p * 3 + 0];
    float y = xb[p * 3 + 1];
    float z = xb[p * 3 + 2];
    sx[p] = make_float4(x, y, z, x * x + y * y + z * z);
  }
  __syncthreads();

  const int i = blockIdx.x * blockDim.x + threadIdx.x;
  const float4 xi = sx[i];

  float bd[KLIST];
  int   bidx[KLIST];
#pragma unroll
  for (int k = 0; k < KLIST; ++k) { bd[k] = __builtin_inff(); bidx[k] = 0x7fffffff; }

  for (int j = 0; j < NPOINTS; ++j) {
    float4 xj = sx[j];
    // Same formula as the reference: sq_i + sq_j - 2*dot, clamped at 0.
    float d2 = xi.w + xj.w - 2.0f * (xi.x * xj.x + xi.y * xj.y + xi.z * xj.z);
    d2 = fmaxf(d2, 0.0f);
    if (d2 < bd[KLIST - 1] || (d2 == bd[KLIST - 1] && j < bidx[KLIST - 1])) {
      float cd = d2; int ci = j;
#pragma unroll
      for (int k = 0; k < KLIST; ++k) {
        bool better = (cd < bd[k]) || (cd == bd[k] && ci < bidx[k]);
        float td = bd[k]; int ti = bidx[k];
        bd[k]   = better ? cd : td;
        bidx[k] = better ? ci : ti;
        cd = better ? td : cd;
        ci = better ? ti : ci;
      }
    }
  }

  int* outp = nbr + ((size_t)b * NPOINTS + i) * NNBR;
#pragma unroll
  for (int k = 1; k < KLIST; ++k) outp[k - 1] = bidx[k];  // drop self (slot 0)
}

// ---------------------------------------------------------------------------
// Kernel 2: one wave32 per point. A = 16x32 f16 (16 neighbor rows, K=12 feats
// zero-padded to 32), B = 32x16 f16 slice of W (4 tiles cover DOUT=64),
// C preloaded with bias so D = feats*W + b; then ReLU, sum over the 16 rows,
// * 1/16.
//
// A layout (ISA 16-bit A 16x32): lane L<16 -> row L, elems 0..7 = K0..7;
// lane L>=16 -> row L-16, elems 0..7 = K8..15. K>=12 padded zero.
// B layout (dense 32x16): lane L<16 -> col L, elems 0..7 = K0..7;
// lane L>=16 -> col L-16, elems 0..7 = K8..15 (same k order as A).
// D layout: value(M,N) with N = lane&15, M = vgpr + 8*(lane>=16).
// ---------------------------------------------------------------------------
__global__ __launch_bounds__(256) void mlp_kernel(const float* __restrict__ x0,
                                                  const float* __restrict__ v0,
                                                  const float* __restrict__ W,
                                                  const float* __restrict__ bias,
                                                  const int* __restrict__ nbr,
                                                  float* __restrict__ out) {
  const int lane = threadIdx.x & 31;
  const int wave = (blockIdx.x * blockDim.x + threadIdx.x) >> 5;
  const int b = wave >> 12;            // / NPOINTS
  const int p = wave & (NPOINTS - 1);  // % NPOINTS
  const size_t base = (size_t)b * NPOINTS;
  const float* xb = x0 + base * 3;
  const float* vb = v0 + base * 3;

  const float xix = xb[p * 3 + 0], xiy = xb[p * 3 + 1], xiz = xb[p * 3 + 2];
  const float vix = vb[p * 3 + 0], viy = vb[p * 3 + 1], viz = vb[p * 3 + 2];

  const int row = lane & 15;  // neighbor slot this lane contributes
  const int j = nbr[(base + p) * NNBR + row];
  const float xjx = xb[j * 3 + 0], xjy = xb[j * 3 + 1], xjz = xb[j * 3 + 2];
  const float vjx = vb[j * 3 + 0], vjy = vb[j * 3 + 1], vjz = vb[j * 3 + 2];

  const bool lo = lane < 16;

  // feats = [Xi(3), Xj(3), Vi(3), Vj(3)]; low lanes carry K0..7, high K8..11.
  v16h a = {};
  if (lo) {
    a[0] = (_Float16)xix; a[1] = (_Float16)xiy; a[2] = (_Float16)xiz;
    a[3] = (_Float16)xjx; a[4] = (_Float16)xjy; a[5] = (_Float16)xjz;
    a[6] = (_Float16)vix; a[7] = (_Float16)viy;
  } else {
    a[0] = (_Float16)viz;
    a[1] = (_Float16)vjx; a[2] = (_Float16)vjy; a[3] = (_Float16)vjz;
  }

  float* outp = out + (base + p) * DOUT;

#pragma unroll
  for (int t = 0; t < 4; ++t) {
    const int col = t * 16 + row;

    v16h wv = {};
    if (lo) {
#pragma unroll
      for (int k = 0; k < 8; ++k) wv[k] = (_Float16)W[k * DOUT + col];
    } else {
#pragma unroll
      for (int k = 0; k < 4; ++k) wv[k] = (_Float16)W[(8 + k) * DOUT + col];
    }  // lanes >=16 hold K8..11 in elems 0..3; K12..31 = 0

    const float bv = bias[col];
    v8f c;
#pragma unroll
    for (int v = 0; v < 8; ++v) c[v] = bv;  // bias into every (M,N) accumulator

    // EXEC is full here (divergent setup reconverged above).
    c = __builtin_amdgcn_wmma_f32_16x16x32_f16(
        /*neg_a=*/false, a, /*neg_b=*/false, wv,
        /*c_mod=*/(short)0, c, /*reuse_a=*/false, /*reuse_b=*/false);

    float s = 0.0f;
#pragma unroll
    for (int v = 0; v < 8; ++v) s += fmaxf(c[v], 0.0f);  // ReLU + partial row sum
    s += __shfl_xor(s, 16, 32);                          // fold M halves

    if (lo) outp[col] = s * (1.0f / 16.0f);
  }
}

// ---------------------------------------------------------------------------
extern "C" void kernel_launch(void* const* d_in, const int* in_sizes, int n_in,
                              void* d_out, int out_size, void* d_ws, size_t ws_size,
                              hipStream_t stream) {
  const float* x0   = (const float*)d_in[0];
  const float* v0   = (const float*)d_in[1];
  const float* W    = (const float*)d_in[2];
  const float* bias = (const float*)d_in[3];

  const int Bn = in_sizes[0] / (NPOINTS * 3);

  int* nbr = (int*)d_ws;  // Bn * NPOINTS * 16 ints (1 MB for Bn=4)

  dim3 g1(NPOINTS / 256, Bn);
  knn_kernel<<<g1, 256, 0, stream>>>(x0, nbr);

  const int waves = Bn * NPOINTS;              // one wave32 per point
  mlp_kernel<<<waves / 8, 256, 0, stream>>>(x0, v0, W, bias, nbr, (float*)d_out);

  // Second tuple element: pass v0 through.
  hipMemcpyAsync((float*)d_out + (size_t)Bn * NPOINTS * DOUT, v0,
                 (size_t)Bn * NPOINTS * 3 * sizeof(float),
                 hipMemcpyDeviceToDevice, stream);
}
